// Attention_62362925138371
// MI455X (gfx1250) — compile-verified, matrix operand order checked by reading, not run on hardware
//
#include <hip/hip_runtime.h>
#include <hip/hip_bf16.h>

#define DIMX   1024
#define DH     64
#define HEADS  16
#define NB     4
#define SEQ    1024
#define NROWS  (NB*SEQ)          // 4096
#define SCALEF 0.125f            // 64^-0.5
#define LN_EPS 1e-5f

typedef __attribute__((ext_vector_type(16))) __bf16 v16bf;
typedef __attribute__((ext_vector_type(8)))  __bf16 v8bf;
typedef __attribute__((ext_vector_type(8)))  float  v8f;

#define WMMA_BF16(a,b,c) \
  __builtin_amdgcn_wmma_f32_16x16x32_bf16(false,(a),false,(b),(short)0,(c),false,false)

static __device__ __forceinline__ unsigned short f2bf(float f) {
  union { float f; unsigned int u; } v; v.f = f;
  unsigned int r = ((v.u >> 16) & 1u) + 0x7FFFu;
  return (unsigned short)((v.u + r) >> 16);
}

// Load a 16x32 bf16 A/B^T fragment (row-major tile, leading dim ld halfs).
// Per ISA layout: lanes 0-15 hold row L, K = 0..7 and 16..23;
// lanes 16-31 hold row L-16, K = 8..15 and 24..31.
static __device__ __forceinline__ v16bf load_frag(const __bf16* tile, int ld, int lane) {
  const int row = lane & 15;
  const int kb  = (lane >> 4) << 3;
  const v8bf lo = *(const v8bf*)(tile + row * ld + kb);
  const v8bf hi = *(const v8bf*)(tile + row * ld + kb + 16);
  return __builtin_shufflevector(lo, hi, 0,1,2,3,4,5,6,7,8,9,10,11,12,13,14,15);
}

// ---------------- f32 -> bf16 convert ----------------
__global__ void k_cvt(const float* __restrict__ in, unsigned short* __restrict__ out, int n) {
  int i = blockIdx.x * blockDim.x + threadIdx.x;
  if (i < n) out[i] = f2bf(in[i]);
}

// ---------------- QKV projection GEMM ----------------
// C[4096x3072] = X[4096x1024] * Wqkv^T, 32x64 register tile per wave.
// Each 64-wide column tile is exactly one (q/k/v, head) pair.
__global__ void __launch_bounds__(512) k_qkv_gemm(
    const unsigned short* __restrict__ Xus, const unsigned short* __restrict__ Wus,
    unsigned short* __restrict__ qb, unsigned short* __restrict__ kb,
    unsigned short* __restrict__ vT) {
  const __bf16* X = (const __bf16*)Xus;
  const __bf16* W = (const __bf16*)Wus;
  const int lane = threadIdx.x & 31;
  const int wave = threadIdx.x >> 5;
  const int tile = blockIdx.x * 16 + wave;       // 128 row-tiles * 48 col-tiles
  const int tr = tile / 48;                      // 32-row tile
  const int tc = tile % 48;                      // 64-col tile
  const __bf16* xrow = X + tr * 32 * DIMX;
  const __bf16* wrow = W + tc * 64 * DIMX;
  v8f acc[2][4];
  for (int rt = 0; rt < 2; ++rt)
    for (int c = 0; c < 4; ++c) acc[rt][c] = (v8f){};
  for (int kk = 0; kk < DIMX; kk += 32) {
    v16bf a0 = load_frag(xrow + kk, DIMX, lane);
    v16bf a1 = load_frag(xrow + 16 * DIMX + kk, DIMX, lane);
#pragma unroll
    for (int c = 0; c < 4; ++c) {
      v16bf bfr = load_frag(wrow + c * 16 * DIMX + kk, DIMX, lane);
      acc[0][c] = WMMA_BF16(a0, bfr, acc[0][c]);
      acc[1][c] = WMMA_BF16(a1, bfr, acc[1][c]);
    }
  }
  const int e0    = tc * 64;
  const int three = e0 >> 10;            // 0:q 1:k 2:v
  const int h     = (e0 & 1023) >> 6;
  const int col   = lane & 15;
  const int rb    = (lane >> 4) * 8;
  for (int rt = 0; rt < 2; ++rt)
    for (int c = 0; c < 4; ++c)
      for (int r = 0; r < 8; ++r) {
        int grow = tr * 32 + rt * 16 + rb + r;   // 0..4095
        int bb = grow >> 10, ii = grow & 1023;
        int d  = c * 16 + col;
        unsigned short val = f2bf(acc[rt][c][r]);
        if (three == 0)
          qb[(((bb * HEADS + h) * SEQ) + ii) * DH + d] = val;
        else if (three == 1)
          kb[(((bb * HEADS + h) * SEQ) + ii) * DH + d] = val;
        else
          vT[(((bb * HEADS + h) * DH) + d) * SEQ + ii] = val;
      }
}

// ---------------- softmax row stats (pass 1) ----------------
__global__ void __launch_bounds__(512) k_stats(
    const unsigned short* __restrict__ qus, const unsigned short* __restrict__ kus,
    float* __restrict__ mbuf, float* __restrict__ lbuf) {
  const __bf16* Q = (const __bf16*)qus;
  const __bf16* K = (const __bf16*)kus;
  const int lane = threadIdx.x & 31;
  const int wave = threadIdx.x >> 5;
  const int wid = blockIdx.x * 16 + wave;        // 0..4095
  const int it = wid & 63;
  const int bh = wid >> 6;
  const __bf16* qt = Q + (bh * SEQ + it * 16) * DH;
  const __bf16* kh = K + bh * SEQ * DH;
  v16bf a0 = load_frag(qt, DH, lane);
  v16bf a1 = load_frag(qt + 32, DH, lane);
  float m[8], l[8];
  for (int r = 0; r < 8; ++r) { m[r] = -3.0e38f; l[r] = 0.f; }
  for (int jt = 0; jt < 64; ++jt) {
    const __bf16* kt = kh + jt * 16 * DH;
    v16bf b0 = load_frag(kt, DH, lane);
    v16bf b1 = load_frag(kt + 32, DH, lane);
    v8f c = {};
    c = WMMA_BF16(a0, b0, c);
    c = WMMA_BF16(a1, b1, c);
    for (int r = 0; r < 8; ++r) {
      float s = c[r] * SCALEF;
      float tm = s;
      for (int off = 1; off < 16; off <<= 1) tm = fmaxf(tm, __shfl_xor(tm, off, 32));
      float mn = fmaxf(m[r], tm);
      float pe = __expf(s - mn);
      for (int off = 1; off < 16; off <<= 1) pe += __shfl_xor(pe, off, 32);
      l[r] = l[r] * __expf(m[r] - mn) + pe;
      m[r] = mn;
    }
  }
  if ((lane & 15) == 0) {
    int rbase = (lane >> 4) * 8;
    for (int r = 0; r < 8; ++r) {
      mbuf[bh * SEQ + it * 16 + rbase + r] = m[r];
      lbuf[bh * SEQ + it * 16 + rbase + r] = l[r];
    }
  }
}

// ---------------- fused re-attention + AV (pass 2) ----------------
// One workgroup per (b, 16-row i-tile); 16 waves = 16 heads.
__global__ void __launch_bounds__(512) k_reattn(
    const unsigned short* __restrict__ qus, const unsigned short* __restrict__ kus,
    const unsigned short* __restrict__ vTus,
    const float* __restrict__ mbuf, const float* __restrict__ lbuf,
    const float* __restrict__ w_re, const float* __restrict__ ln_g,
    const float* __restrict__ ln_b, unsigned short* __restrict__ obuf) {
  const __bf16* Q  = (const __bf16*)qus;
  const __bf16* K  = (const __bf16*)kus;
  const __bf16* VT = (const __bf16*)vTus;
  __shared__ __align__(16) float          s_attn[16][16][32];  // [h][i][j]
  __shared__ __align__(16) unsigned short s_mix[16][16][32];   // bf16 [g][i][j]
  __shared__ float s_wre[256];
  const int lane = threadIdx.x & 31;
  const int h    = threadIdx.x >> 5;
  const int it   = blockIdx.x & 63;
  const int b    = blockIdx.x >> 6;
  if (threadIdx.x < 256) s_wre[threadIdx.x] = w_re[threadIdx.x];
  const int bh = b * HEADS + h;
  const __bf16* qt = Q + (bh * SEQ + it * 16) * DH;
  v16bf a0 = load_frag(qt, DH, lane);
  v16bf a1 = load_frag(qt + 32, DH, lane);
  float mi[8], rli[8];
  {
    int rbase = (lane >> 4) * 8;
    for (int r = 0; r < 8; ++r) {
      mi[r]  = mbuf[bh * SEQ + it * 16 + rbase + r];
      rli[r] = 1.0f / lbuf[bh * SEQ + it * 16 + rbase + r];
    }
  }
  v8f o0 = {}, o1 = {}, o2 = {}, o3 = {};
  const int rbase = (lane >> 4) * 8;
  const int col   = lane & 15;
  for (int jt = 0; jt < 32; ++jt) {
    const int j0 = jt * 32;
    // --- scores + softmax normalize, 16x32 slab per head ---
    for (int sub = 0; sub < 2; ++sub) {
      const __bf16* kt = K + (bh * SEQ + j0 + sub * 16) * DH;
      v16bf b0 = load_frag(kt, DH, lane);
      v16bf b1 = load_frag(kt + 32, DH, lane);
      v8f c = {};
      c = WMMA_BF16(a0, b0, c);
      c = WMMA_BF16(a1, b1, c);
      for (int r = 0; r < 8; ++r)
        s_attn[h][rbase + r][sub * 16 + col] = __expf(c[r] * SCALEF - mi[r]) * rli[r];
    }
    __syncthreads();
    // --- head mix (16x16) + LayerNorm over heads, per (i,j) ---
    {
      const int ii = threadIdx.x >> 5;   // wave index = i row
      const int jj = threadIdx.x & 31;
      float av[16];
      for (int hh = 0; hh < 16; ++hh) av[hh] = s_attn[hh][ii][jj];
      float mx[16], mu = 0.f;
      for (int g = 0; g < 16; ++g) {
        float a = 0.f;
        for (int hh = 0; hh < 16; ++hh) a += s_wre[g * 16 + hh] * av[hh];
        mx[g] = a; mu += a;
      }
      mu *= (1.0f / 16.0f);
      float var = 0.f;
      for (int g = 0; g < 16; ++g) { float d = mx[g] - mu; var += d * d; }
      var *= (1.0f / 16.0f);
      float inv = rsqrtf(var + LN_EPS);
      for (int g = 0; g < 16; ++g)
        s_mix[g][ii][jj] = f2bf((mx[g] - mu) * inv * ln_g[g] + ln_b[g]);
    }
    __syncthreads();
    // --- O(16x64) += P(16x32) x V(32x64) ---
    v16bf pa = load_frag((const __bf16*)&s_mix[h][0][0], 32, lane);
    const __bf16* vb = VT + bh * DH * SEQ + j0;    // rows = d, ld = SEQ
    o0 = WMMA_BF16(pa, load_frag(vb + 0 * 16 * SEQ, SEQ, lane), o0);
    o1 = WMMA_BF16(pa, load_frag(vb + 1 * 16 * SEQ, SEQ, lane), o1);
    o2 = WMMA_BF16(pa, load_frag(vb + 2 * 16 * SEQ, SEQ, lane), o2);
    o3 = WMMA_BF16(pa, load_frag(vb + 3 * 16 * SEQ, SEQ, lane), o3);
    __syncthreads();
  }
  for (int r = 0; r < 8; ++r) {
    int grow = b * SEQ + it * 16 + rbase + r;
    unsigned short* dst = obuf + grow * DIMX + h * DH;
    dst[0 * 16 + col] = f2bf(o0[r]);
    dst[1 * 16 + col] = f2bf(o1[r]);
    dst[2 * 16 + col] = f2bf(o2[r]);
    dst[3 * 16 + col] = f2bf(o3[r]);
  }
}

// ---------------- output projection GEMM + bias ----------------
// C[4096x1024] = O[4096x1024] * Wout^T + b, 32x64 register tile per wave.
__global__ void __launch_bounds__(512) k_out_gemm(
    const unsigned short* __restrict__ Ous, const unsigned short* __restrict__ Wus,
    const float* __restrict__ bias, float* __restrict__ out) {
  const __bf16* O = (const __bf16*)Ous;
  const __bf16* W = (const __bf16*)Wus;
  const int lane = threadIdx.x & 31;
  const int wave = threadIdx.x >> 5;
  const int tile = blockIdx.x * 16 + wave;       // 128 row-tiles * 16 col-tiles
  const int tr = tile >> 4;
  const int tc = tile & 15;
  const __bf16* orow = O + tr * 32 * DIMX;
  const __bf16* wrow = W + tc * 64 * DIMX;
  v8f acc[2][4];
  for (int rt = 0; rt < 2; ++rt)
    for (int c = 0; c < 4; ++c) acc[rt][c] = (v8f){};
  for (int kk = 0; kk < DIMX; kk += 32) {
    v16bf a0 = load_frag(orow + kk, DIMX, lane);
    v16bf a1 = load_frag(orow + 16 * DIMX + kk, DIMX, lane);
#pragma unroll
    for (int c = 0; c < 4; ++c) {
      v16bf bfr = load_frag(wrow + c * 16 * DIMX + kk, DIMX, lane);
      acc[0][c] = WMMA_BF16(a0, bfr, acc[0][c]);
      acc[1][c] = WMMA_BF16(a1, bfr, acc[1][c]);
    }
  }
  const int rb  = (lane >> 4) * 8;
  const int col = lane & 15;
  for (int rt = 0; rt < 2; ++rt)
    for (int c = 0; c < 4; ++c) {
      float bz = bias[tc * 64 + c * 16 + col];
      for (int r = 0; r < 8; ++r)
        out[(tr * 32 + rt * 16 + rb + r) * DIMX + tc * 64 + c * 16 + col] =
            acc[rt][c][r] + bz;
    }
}

extern "C" void kernel_launch(void* const* d_in, const int* in_sizes, int n_in,
                              void* d_out, int out_size, void* d_ws, size_t ws_size,
                              hipStream_t stream) {
  const float* x     = (const float*)d_in[0];   // [4,1024,1024]
  const float* w_qkv = (const float*)d_in[1];   // [3072,1024]
  const float* w_re  = (const float*)d_in[2];   // [16,16]
  const float* ln_g  = (const float*)d_in[3];   // [16]
  const float* ln_b  = (const float*)d_in[4];   // [16]
  const float* w_out = (const float*)d_in[5];   // [1024,1024]
  const float* b_out = (const float*)d_in[6];   // [1024]
  float* out = (float*)d_out;

  char* ws = (char*)d_ws;
  unsigned short* xbf  = (unsigned short*)(ws);                 // 8 MB
  unsigned short* wqbf = (unsigned short*)(ws + (8u << 20));    // 6 MB
  unsigned short* wobf = (unsigned short*)(ws + (14u << 20));   // 2 MB
  unsigned short* qbuf = (unsigned short*)(ws + (16u << 20));   // 8 MB
  unsigned short* kbuf = (unsigned short*)(ws + (24u << 20));   // 8 MB
  unsigned short* vT   = (unsigned short*)(ws + (32u << 20));   // 8 MB
  float* mbuf          = (float*)(ws + (40u << 20));            // 256 KB
  float* lbuf          = (float*)(ws + (40u << 20) + (256u << 10));
  unsigned short* obuf = (unsigned short*)(ws + (41u << 20));   // 8 MB

  // f32 -> bf16 conversions
  k_cvt<<<(NROWS * DIMX + 511) / 512, 512, 0, stream>>>(x, xbf, NROWS * DIMX);
  k_cvt<<<(3 * DIMX * DIMX + 511) / 512, 512, 0, stream>>>(w_qkv, wqbf, 3 * DIMX * DIMX);
  k_cvt<<<(DIMX * DIMX + 511) / 512, 512, 0, stream>>>(w_out, wobf, DIMX * DIMX);

  // QKV projection: 128 x 48 (32x64) tiles, 16 waves per block
  k_qkv_gemm<<<(128 * 48) / 16, 512, 0, stream>>>(xbf, wqbf, qbuf, kbuf, vT);

  // softmax stats: 4096 waves
  k_stats<<<4096 / 16, 512, 0, stream>>>(qbuf, kbuf, mbuf, lbuf);

  // fused re-attention: one block per (b, i-tile)
  k_reattn<<<NB * (SEQ / 16), 512, 0, stream>>>(qbuf, kbuf, vT, mbuf, lbuf,
                                                w_re, ln_g, ln_b, obuf);

  // output projection: 128 x 16 (32x64) tiles
  k_out_gemm<<<(128 * 16) / 16, 512, 0, stream>>>(obuf, wobf, b_out, out);
}
